// PAGNet_encoding_20942260535658
// MI455X (gfx1250) — compile-verified, matrix operand order checked by reading, not run on hardware
//
#include <hip/hip_runtime.h>
#include <hip/hip_bf16.h>

// ---------------- problem constants ----------------
#define BB 4
#define NPT 16384
#define MPTS (BB*NPT)      // 65536 points
#define NSEL 4096
#define HF 352
#define WF 400
#define HWF (HF*WF)        // 140800
#define HH 176
#define WH 200
#define HWH (HH*WH)        // 35200

typedef __attribute__((ext_vector_type(16))) _Float16 v16h;
typedef __attribute__((ext_vector_type(8)))  _Float16 v8h;
typedef __attribute__((ext_vector_type(8)))  float    v8f;

// ---------------- per-point MLP: 4->16->16->32 (tiny; VALU) ----------------
__global__ void point_mlp_kernel(const float* __restrict__ pts,
                                 const float* __restrict__ w1,
                                 const float* __restrict__ w2,
                                 const float* __restrict__ w3,
                                 float* __restrict__ pw) {
  int p = blockIdx.x * blockDim.x + threadIdx.x;
  if (p >= MPTS) return;
  float x[4];
#pragma unroll
  for (int i = 0; i < 4; ++i) x[i] = pts[p * 5 + 1 + i];
  float h1[16];
#pragma unroll
  for (int n = 0; n < 16; ++n) {
    float s = 0.f;
#pragma unroll
    for (int i = 0; i < 4; ++i) s += x[i] * w1[i * 16 + n];
    h1[n] = fmaxf(s, 0.f);
  }
  float h2[16];
#pragma unroll
  for (int n = 0; n < 16; ++n) {
    float s = 0.f;
#pragma unroll
    for (int i = 0; i < 16; ++i) s += h1[i] * w2[i * 16 + n];
    h2[n] = fmaxf(s, 0.f);
  }
#pragma unroll
  for (int n = 0; n < 32; ++n) {
    float s = 0.f;
#pragma unroll
    for (int i = 0; i < 16; ++i) s += h2[i] * w3[i * 32 + n];
    pw[(long)p * 32 + n] = fmaxf(s, 0.f);
  }
}

// ---------------- BEV flat index ----------------
__device__ __forceinline__ int bev_flat(float bx, float x, float y) {
  int b = (int)bx;
  int u = (int)floorf(x * 5.0f);            // (x - 0.0)/0.2
  int v = (int)floorf((y + 40.0f) * 5.0f);  // (y + 40)/0.2
  u = min(max(u, 0), HF - 1);
  v = min(max(v, 0), WF - 1);
  return b * HWF + u * WF + v;
}

__global__ void flat_idx_kernel(const float* __restrict__ pts, int* __restrict__ flat, int n) {
  int p = blockIdx.x * blockDim.x + threadIdx.x;
  if (p >= n) return;
  flat[p] = bev_flat(pts[p * 5 + 0], pts[p * 5 + 1], pts[p * 5 + 2]);
}

// ---------------- scatter-mean (p2g) ----------------
__global__ void scatter_add_kernel(const float* __restrict__ feat, const int* __restrict__ flat,
                                   float* __restrict__ sum, float* __restrict__ cnt,
                                   int n, int C) {
  int p = blockIdx.x * blockDim.x + threadIdx.x;
  if (p >= n) return;
  int f = flat[p];
  int b = f / HWF, hw = f - b * HWF;
  for (int c = 0; c < C; ++c)
    atomicAdd(&sum[((long)(b * C + c)) * HWF + hw], feat[(long)p * C + c]);
  atomicAdd(&cnt[f], 1.0f);
}

__global__ void p2g_final_kernel(float* __restrict__ bev, const float* __restrict__ cnt,
                                 int C, const float* __restrict__ addsrc, int addC) {
  long i = (long)blockIdx.x * blockDim.x + threadIdx.x;
  long total = (long)BB * C * HWF;
  if (i >= total) return;
  int hw = (int)(i % HWF);
  int c  = (int)((i / HWF) % C);
  int b  = (int)(i / ((long)HWF * C));
  float d = cnt[b * HWF + hw];
  float v = bev[i] / fmaxf(d, 1.0f);
  if (addsrc && c < addC) v += addsrc[((long)(b * addC + c)) * HWF + hw];
  bev[i] = v;
}

// ---------------- WMMA implicit-GEMM conv, LDS-tiled, K=(tap,ci) order ----------------
// Block = WAVES waves -> (2*WAVES)x8 output pixels x 32 out channels.
// Wave tile = 2 rows x 8 cols (A-row M -> (M>>3, M&7)). K runs (tap, ci) so a
// 32-wide K-chunk never crosses a 3x3 tap (CIN is 32/64); with the input tile
// staged CHANNEL-LAST (lt[(iy*ITX+ix)*CIN+ci]) each lane's A fragment is two
// contiguous 16B ds_load_b128 + shuffle -- same as the B fragment from the
// LDS weight tile (permuted OIHW -> (co,tap,ci) during staging; CIN is a
// power of two so all staging index math is shifts/ands).
// MODE: 0 normal, 1 = fused 2x nearest upsample input, 2 = fused concat [X,X2].
template<int CIN, int KS, int PAD, int STRIDE, int MODE, int WAVES>
__global__ __launch_bounds__(WAVES * 32)
void conv_wmma_t(const float* __restrict__ X, const float* __restrict__ X2,
                 const float* __restrict__ Wt, const float* __restrict__ bias,
                 float* __restrict__ Y, int Cout, int Hv, int Wv, int Hout, int Wout) {
  constexpr int KK2 = KS * KS;
  constexpr int KK  = CIN * KK2;              // 288 / 576 / 32
  constexpr int NCH = KK / 32;
  constexpr int ROWS = 2 * WAVES;
  constexpr int ITY = (ROWS - 1) * STRIDE + KS;   // 10 / 9 / 8
  constexpr int ITX = 7 * STRIDE + KS;            // 10 / 17 / 8
  constexpr int NT  = WAVES * 32;
  constexpr int CSH = (CIN == 64) ? 6 : 5;        // log2(CIN)
  __shared__ _Float16 lt[ITY * ITX * CIN];
  __shared__ _Float16 lw[32 * KK];

  const int tid = threadIdx.x;
  const int wave = tid >> 5, lane = tid & 31;
  const int hi = (lane >> 4) & 1, nm = lane & 15;
  const int ct = blockIdx.y, b = blockIdx.z;

  const int wt8 = Wout >> 3;
  const int by = blockIdx.x / wt8;                // one scalar div per block
  const int bx = blockIdx.x - by * wt8;
  const int oy0 = by * ROWS, ox0 = bx * 8;

  // ---- stage input halo tile, channel-last ----
  for (int i = tid; i < CIN * ITY * ITX; i += NT) {
    const int ci  = i / (ITY * ITX);              // global-coalesced order
    const int pos = i - ci * (ITY * ITX);
    const int iyl = pos / ITX, ixl = pos - iyl * ITX;
    const int gy = oy0 * STRIDE - PAD + iyl;
    const int gx = ox0 * STRIDE - PAD + ixl;
    float v = 0.f;
    if (gy >= 0 && gy < Hv && gx >= 0 && gx < Wv) {
      if (MODE == 1) {
        v = X[(((long)b * CIN + ci) * (Hv >> 1) + (gy >> 1)) * (Wv >> 1) + (gx >> 1)];
      } else if (MODE == 2) {
        const float* s = (ci < CIN / 2) ? X : X2;
        const int cc   = (ci < CIN / 2) ? ci : ci - CIN / 2;
        v = s[(((long)b * (CIN / 2) + cc) * Hv + gy) * Wv + gx];
      } else {
        v = X[(((long)b * CIN + ci) * Hv + gy) * Wv + gx];
      }
    }
    lt[(pos << CSH) + ci] = (_Float16)v;
  }
  // ---- stage weight tile permuted to (co, tap, ci) ----
  const float* wsrc = Wt + (size_t)(ct * 32) * KK;
  for (int i = tid; i < 32 * KK; i += NT) {
    const int cl  = i / KK;
    const int kk  = i - cl * KK;
    const int tap = kk >> CSH;
    const int ci  = kk & (CIN - 1);
    lw[i] = (_Float16)wsrc[(cl * CIN + ci) * KK2 + tap];
  }
  __syncthreads();

  const int my = nm >> 3, mx = nm & 7;            // A-row -> pixel in tile
  const int co0 = ct * 32 + nm;

  v8f acc0 = {}, acc1 = {};
#pragma unroll
  for (int kc = 0; kc < NCH; ++kc) {
    const int tap = (kc * 32) / CIN;              // compile-time per iteration
    const int cib = kc * 32 - tap * CIN;          // 0 or 32
    const int ky = tap / KS, kx = tap - ky * KS;
    const int arow = ((wave * 2 + my) * STRIDE + ky) * ITX + (mx * STRIDE + kx);
    const _Float16* ap = &lt[(arow << CSH) + cib + (hi ? 8 : 0)];
    v8h alo = *(const v8h*)ap, ahi = *(const v8h*)(ap + 16);
    v16h a = __builtin_shufflevector(alo, ahi, 0,1,2,3,4,5,6,7,8,9,10,11,12,13,14,15);

    const int kb = kc * 32 + (hi ? 8 : 0);
    const _Float16* w0p = &lw[nm * KK + kb];       // 16B aligned (KK % 8 == 0)
    const _Float16* w1p = &lw[(nm + 16) * KK + kb];
    v8h w0lo = *(const v8h*)w0p, w0hi = *(const v8h*)(w0p + 16);
    v8h w1lo = *(const v8h*)w1p, w1hi = *(const v8h*)(w1p + 16);
    v16h w0 = __builtin_shufflevector(w0lo, w0hi, 0,1,2,3,4,5,6,7,8,9,10,11,12,13,14,15);
    v16h w1 = __builtin_shufflevector(w1lo, w1hi, 0,1,2,3,4,5,6,7,8,9,10,11,12,13,14,15);
    acc0 = __builtin_amdgcn_wmma_f32_16x16x32_f16(false, a, false, w0, (short)0, acc0, false, false);
    acc1 = __builtin_amdgcn_wmma_f32_16x16x32_f16(false, a, false, w1, (short)0, acc1, false, false);
  }
  // ---- store: C-row M = r + 8*hi -> (row hi, col r): 8 contiguous floats ----
  const float bv0 = bias[co0], bv1 = bias[co0 + 16];
  const int oy2 = oy0 + wave * 2 + hi;
  float* y0 = &Y[(((long)b * Cout + co0) * Hout + oy2) * Wout + ox0];
  float* y1 = &Y[(((long)b * Cout + co0 + 16) * Hout + oy2) * Wout + ox0];
#pragma unroll
  for (int r = 0; r < 8; ++r) { y0[r] = acc0[r] + bv0; y1[r] = acc1[r] + bv1; }
}

// ---------------- dense GEMM via WMMA: Y[M,N] = relu(X[M,K] @ W[K,N]) ----------------
template<int K>
__global__ __launch_bounds__(128)
void gemm_wmma_t(const float* __restrict__ X, const float* __restrict__ Wt,
                 float* __restrict__ Y, int N, int relu_f) {
  __shared__ _Float16 lw[32 * K];
  const int tid = threadIdx.x, wave = tid >> 5, lane = tid & 31;
  const int hi = lane >> 4, nm = lane & 15;
  const int col0 = blockIdx.y * 32;
  // stage transposed so fragment reads are contiguous: lw[c*K + k] = W[k][col0+c]
  for (int i = tid; i < 32 * K; i += 128) {
    int c = i / K, k = i - (i / K) * K;
    lw[i] = (_Float16)Wt[(long)k * N + col0 + c];
  }
  __syncthreads();
  const long row = (long)(blockIdx.x * 4 + wave) * 16;
  v8f acc0 = {}, acc1 = {};
#pragma unroll
  for (int kc = 0; kc < K / 32; ++kc) {
    v16h a;
#pragma unroll
    for (int j = 0; j < 16; ++j) {
      int k = kc * 32 + (hi ? 8 : 0) + (j < 8 ? j : j + 8);
      a[j] = (_Float16)X[(row + nm) * K + k];
    }
    const int kb = kc * 32 + (hi ? 8 : 0);
    const _Float16* w0p = &lw[nm * K + kb];
    const _Float16* w1p = &lw[(nm + 16) * K + kb];
    v8h w0lo = *(const v8h*)w0p, w0hi = *(const v8h*)(w0p + 16);
    v8h w1lo = *(const v8h*)w1p, w1hi = *(const v8h*)(w1p + 16);
    v16h w0 = __builtin_shufflevector(w0lo, w0hi, 0,1,2,3,4,5,6,7,8,9,10,11,12,13,14,15);
    v16h w1 = __builtin_shufflevector(w1lo, w1hi, 0,1,2,3,4,5,6,7,8,9,10,11,12,13,14,15);
    acc0 = __builtin_amdgcn_wmma_f32_16x16x32_f16(false, a, false, w0, (short)0, acc0, false, false);
    acc1 = __builtin_amdgcn_wmma_f32_16x16x32_f16(false, a, false, w1, (short)0, acc1, false, false);
  }
#pragma unroll
  for (int r = 0; r < 8; ++r) {
    int mm = r + (hi ? 8 : 0);
    float v0 = acc0[r], v1 = acc1[r];
    if (relu_f) { v0 = fmaxf(v0, 0.f); v1 = fmaxf(v1, 0.f); }
    Y[(row + mm) * N + col0 + nm]      = v0;
    Y[(row + mm) * N + col0 + 16 + nm] = v1;
  }
}

// ---------------- batch-norm: stats then apply ----------------
__global__ void bn_stats_kernel(const float* __restrict__ X, float* __restrict__ stats,
                                int C, int HoWo) {
  int c = blockIdx.x;
  __shared__ float ss[256], sq[256];
  float s = 0.f, q = 0.f;
  long total = (long)BB * HoWo;
  for (long i = threadIdx.x + (long)blockIdx.y * blockDim.x; i < total;
       i += (long)gridDim.y * blockDim.x) {
    int b  = (int)(i / HoWo);
    int hw = (int)(i - (long)b * HoWo);
    float v = X[((long)(b * C + c)) * HoWo + hw];
    s += v; q += v * v;
  }
  ss[threadIdx.x] = s; sq[threadIdx.x] = q;
  __syncthreads();
  for (int o = 128; o > 0; o >>= 1) {
    if ((int)threadIdx.x < o) { ss[threadIdx.x] += ss[threadIdx.x + o]; sq[threadIdx.x] += sq[threadIdx.x + o]; }
    __syncthreads();
  }
  if (threadIdx.x == 0) { atomicAdd(&stats[c], ss[0]); atomicAdd(&stats[C + c], sq[0]); }
}

__global__ void bn_apply_kernel(float* __restrict__ Y, const float* __restrict__ X,
                                const float* __restrict__ stats,
                                const float* __restrict__ g, const float* __restrict__ bb,
                                int C, int HoWo, float inv_cnt, int relu_f,
                                const float* __restrict__ resid) {
  long i = (long)blockIdx.x * blockDim.x + threadIdx.x;
  long total = (long)BB * C * HoWo;
  if (i >= total) return;
  int c = (int)((i / HoWo) % C);
  float mean = stats[c] * inv_cnt;
  float var  = stats[C + c] * inv_cnt - mean * mean;
  float v = (X[i] - mean) * rsqrtf(var + 1e-5f) * g[c] + bb[c];
  if (resid) v += resid[i];
  if (relu_f) v = fmaxf(v, 0.f);
  Y[i] = v;
}

// ---------------- maxpool 2x2 ----------------
__global__ void maxpool2_kernel(const float* __restrict__ X, float* __restrict__ Y,
                                int C, int Hi, int Wi) {
  int Ho = Hi >> 1, Wo = Wi >> 1;
  long i = (long)blockIdx.x * blockDim.x + threadIdx.x;
  long total = (long)BB * C * Ho * Wo;
  if (i >= total) return;
  int ox = (int)(i % Wo);
  int oy = (int)((i / Wo) % Ho);
  long bc = i / ((long)Ho * Wo);
  const float* src = X + bc * Hi * Wi;
  float a = src[(2 * oy) * Wi + 2 * ox];
  float b = src[(2 * oy) * Wi + 2 * ox + 1];
  float c = src[(2 * oy + 1) * Wi + 2 * ox];
  float d = src[(2 * oy + 1) * Wi + 2 * ox + 1];
  Y[i] = fmaxf(fmaxf(a, b), fmaxf(c, d));
}

// ---------------- g2p gather + concat [g, pw] ----------------
__global__ void g2p_concat_kernel(const float* __restrict__ bev, const int* __restrict__ flat,
                                  const float* __restrict__ pw, float* __restrict__ cmplt) {
  int p = blockIdx.x * blockDim.x + threadIdx.x;
  if (p >= MPTS) return;
  int f = flat[p];
  int b = f / HWF, hw = f - b * HWF;
  for (int c = 0; c < 32; ++c)
    cmplt[(long)p * 64 + c] = bev[((long)(b * 32 + c)) * HWF + hw];
  const float4* src = (const float4*)(pw + (long)p * 32);
  float4* dst = (float4*)(cmplt + (long)p * 64 + 32);
#pragma unroll
  for (int c = 0; c < 8; ++c) dst[c] = src[c];
}

// ---------------- sem head: [M,64] @ cw3[64,4] ----------------
__global__ void sem_head_kernel(const float* __restrict__ s2, const float* __restrict__ cw3,
                                float* __restrict__ sem) {
  int p = blockIdx.x * blockDim.x + threadIdx.x;
  if (p >= MPTS) return;
  float o[4] = {0.f, 0.f, 0.f, 0.f};
  for (int i = 0; i < 64; ++i) {
    float v = s2[(long)p * 64 + i];
#pragma unroll
    for (int c = 0; c < 4; ++c) o[c] += v * cw3[i * 4 + c];
  }
#pragma unroll
  for (int c = 0; c < 4; ++c) sem[(long)p * 4 + c] = o[c];
}

// ---------------- fg-priority score ----------------
__global__ void score_kernel(const float* __restrict__ sem, float* __restrict__ score) {
  int p = blockIdx.x * blockDim.x + threadIdx.x;
  if (p >= MPTS) return;
  float s0 = sem[(long)p * 4 + 0], s1 = sem[(long)p * 4 + 1];
  float s2 = sem[(long)p * 4 + 2], s3 = sem[(long)p * 4 + 3];
  float mx = fmaxf(s1, fmaxf(s2, s3));
  float sc;
  if (mx > s0) {                      // argmax > 0
    sc = 1.f / (1.f + expf(-mx)) + 2.0f;
  } else {
    float g0 = 1.f / (1.f + expf(-s0)), g1 = 1.f / (1.f + expf(-s1));
    float g2 = 1.f / (1.f + expf(-s2)), g3 = 1.f / (1.f + expf(-s3));
    float nrm = sqrtf(g0 * g0 + g1 * g1 + g2 * g2 + g3 * g3);
    sc = -(g0 / nrm);
  }
  score[p] = sc;
}

// ---------------- per-batch top-4096: bitonic sort of 16384 64-bit keys in LDS ----------------
__global__ void topk_kernel(const float* __restrict__ score, int* __restrict__ selrow) {
  extern __shared__ unsigned long long skey[];   // 16384 * 8B = 128 KB (320 KB WGP LDS)
  int b = blockIdx.x;
  int tid = threadIdx.x, nt = blockDim.x;
  for (int i = tid; i < NPT; i += nt) {
    unsigned u = __float_as_uint(score[b * NPT + i]);
    u = (u & 0x80000000u) ? ~u : (u | 0x80000000u);   // order-preserving key
    skey[i] = ((unsigned long long)u << 32) | (unsigned)(NPT - 1 - i);  // low idx wins
  }
  __syncthreads();
  for (int k = 2; k <= NPT; k <<= 1) {
    for (int j = k >> 1; j > 0; j >>= 1) {
      for (int i = tid; i < NPT; i += nt) {
        int ixj = i ^ j;
        if (ixj > i) {
          bool desc = ((i & k) == 0);
          unsigned long long a = skey[i], c = skey[ixj];
          bool sw = desc ? (a < c) : (a > c);
          if (sw) { skey[i] = c; skey[ixj] = a; }
        }
      }
      __syncthreads();
    }
  }
  for (int i = tid; i < NSEL; i += nt) {
    int idx = NPT - 1 - (int)(skey[i] & 0xFFFFFFFFu);
    selrow[b * NSEL + i] = b * NPT + idx;
  }
}

// ---------------- gather selected points / feats, compute new flat ----------------
__global__ void gather_sel_kernel(const int* __restrict__ selrow, const float* __restrict__ pts,
                                  const float* __restrict__ cmplt,
                                  float* __restrict__ nfeat, int* __restrict__ nflat) {
  int i = blockIdx.x * blockDim.x + threadIdx.x;
  if (i >= BB * NSEL) return;
  int row = selrow[i];
  const float4* src = (const float4*)(cmplt + (long)row * 64);
  float4* dst = (float4*)(nfeat + (long)i * 64);
#pragma unroll
  for (int c = 0; c < 16; ++c) dst[c] = src[c];
  nflat[i] = bev_flat(pts[row * 5 + 0], pts[row * 5 + 1], pts[row * 5 + 2]);
}

// ================== host orchestration ==================
extern "C" void kernel_launch(void* const* d_in, const int* in_sizes, int n_in,
                              void* d_out, int out_size, void* d_ws, size_t ws_size,
                              hipStream_t stream) {
  const float* pts = (const float*)d_in[0];
  const float* w1 = (const float*)d_in[2];
  const float* w2 = (const float*)d_in[3];
  const float* w3 = (const float*)d_in[4];
  const float* cw1 = (const float*)d_in[5];
  const float* cw2 = (const float*)d_in[6];
  const float* cw3 = (const float*)d_in[7];
  const float* u11w = (const float*)d_in[8],  *u11b = (const float*)d_in[9],
             * u11g = (const float*)d_in[10], *u11bb = (const float*)d_in[11];
  const float* u12w = (const float*)d_in[12], *u12b = (const float*)d_in[13],
             * u12g = (const float*)d_in[14], *u12bb = (const float*)d_in[15];
  const float* u21w = (const float*)d_in[16], *u21b = (const float*)d_in[17],
             * u21g = (const float*)d_in[18], *u21bb = (const float*)d_in[19];
  const float* u22w = (const float*)d_in[20], *u22b = (const float*)d_in[21],
             * u22g = (const float*)d_in[22], *u22bb = (const float*)d_in[23];
  const float* uupw = (const float*)d_in[24], *uupb = (const float*)d_in[25],
             * uupg = (const float*)d_in[26], *uupbb = (const float*)d_in[27];
  const float* ud1w = (const float*)d_in[28], *ud1b = (const float*)d_in[29],
             * ud1g = (const float*)d_in[30], *ud1bb = (const float*)d_in[31];
  const float* uow = (const float*)d_in[32],  *uob = (const float*)d_in[33];
  const float* bc1w = (const float*)d_in[34], *bc1b = (const float*)d_in[35],
             * bn1g = (const float*)d_in[36], *bn1b = (const float*)d_in[37];
  const float* bc2w = (const float*)d_in[38], *bc2b = (const float*)d_in[39],
             * bn2g = (const float*)d_in[40], *bn2b = (const float*)d_in[41];
  const float* bc3w = (const float*)d_in[42], *bc3b = (const float*)d_in[43],
             * bn3g = (const float*)d_in[44], *bn3b = (const float*)d_in[45];
  float* out = (float*)d_out;                       // [4,128,176,200]

  // ---- workspace layout (floats) ----
  float* base = (float*)d_ws;
  size_t off = 0;
  float* PW      = base + off; off += (size_t)MPTS * 32;
  float* CNT     = base + off; off += (size_t)BB * HWF;
  float* STATS   = base + off; off += 256;
  float* SEM     = base + off; off += (size_t)MPTS * 4;
  float* SCORE   = base + off; off += MPTS;
  int*   SELROW  = (int*)(base + off); off += BB * NSEL;
  float* NEWFEAT = base + off; off += (size_t)BB * NSEL * 64;
  int*   NFLAT   = (int*)(base + off); off += BB * NSEL;
  int*   FLAT    = (int*)(base + off); off += MPTS;
  float* HALF0   = base + off; off += (size_t)BB * 64 * HWH;
  float* HALF1   = base + off; off += (size_t)BB * 64 * HWH;
  float* BEV0    = base + off; off += (size_t)BB * 64 * HWF;
  float* BEV1    = base + off; off += (size_t)BB * 64 * HWF;
  float* BEV2    = base + off; off += (size_t)BB * 64 * HWF;
  float* CMPLT = BEV0;                     // [M,64] alias (free during MLP phase)
  float* S1 = BEV1;                        // [M,64]
  float* S2 = BEV1 + (size_t)MPTS * 64;    // [M,64]

  const int T = 256;
  auto blk = [](long n, int t) { return dim3((unsigned)((n + t - 1) / t)); };

  auto bnrelu = [&](float* Y, const float* X, const float* g, const float* b,
                    int C, int HoWo, int relu_f, const float* resid) {
    hipMemsetAsync(STATS, 0, 256 * sizeof(float), stream);
    bn_stats_kernel<<<dim3(C, 64), 256, 0, stream>>>(X, STATS, C, HoWo);
    long total = (long)BB * C * HoWo;
    bn_apply_kernel<<<blk(total, T), T, 0, stream>>>(Y, X, STATS, g, b, C, HoWo,
                                                     1.0f / (float)(BB * HoWo), relu_f, resid);
  };
  // conv grids: x = (H/ROWS)*(W/8) blocks, y = Cout/32, z = batch
  const int NBF = (HF / 8) * (WF / 8);      // 2200 full-res blocks (4 waves)
  const int NBH = (HH / 8) * (WH / 8);      // 550 half-res blocks (4 waves)
  const int NBS = (HH / 4) * (WH / 8);      // 1100 stride-2 blocks (2 waves)
  const dim3 gF1(NBF, 1, BB), gF2(NBF, 2, BB);
  const dim3 gH2(NBH, 2, BB), gS4(NBS, 4, BB);

  // 1) per-point MLP + flat index
  point_mlp_kernel<<<blk(MPTS, T), T, 0, stream>>>(pts, w1, w2, w3, PW);
  flat_idx_kernel<<<blk(MPTS, T), T, 0, stream>>>(pts, FLAT, MPTS);

  // 2) p2g scatter-mean -> init_bev (BEV0, 32ch)
  hipMemsetAsync(BEV0, 0, (size_t)BB * 32 * HWF * sizeof(float), stream);
  hipMemsetAsync(CNT, 0, (size_t)BB * HWF * sizeof(float), stream);
  scatter_add_kernel<<<blk(MPTS, T), T, 0, stream>>>(PW, FLAT, BEV0, CNT, MPTS, 32);
  p2g_final_kernel<<<blk((long)BB * 32 * HWF, T), T, 0, stream>>>(BEV0, CNT, 32, nullptr, 0);

  // 3) UNet
  conv_wmma_t<32,3,1,1,0,4><<<gF1, 128, 0, stream>>>(BEV0, nullptr, u11w, u11b, BEV1, 32, HF, WF, HF, WF);
  bnrelu(BEV1, BEV1, u11g, u11bb, 32, HWF, 1, nullptr);
  conv_wmma_t<32,3,1,1,0,4><<<gF1, 128, 0, stream>>>(BEV1, nullptr, u12w, u12b, BEV2, 32, HF, WF, HF, WF);
  bnrelu(BEV2, BEV2, u12g, u12bb, 32, HWF, 1, nullptr);             // e1 in BEV2
  maxpool2_kernel<<<blk((long)BB * 32 * HWH, T), T, 0, stream>>>(BEV2, HALF0, 32, HF, WF);
  conv_wmma_t<32,3,1,1,0,4><<<gH2, 128, 0, stream>>>(HALF0, nullptr, u21w, u21b, HALF1, 64, HH, WH, HH, WH);
  bnrelu(HALF1, HALF1, u21g, u21bb, 64, HWH, 1, nullptr);
  conv_wmma_t<64,3,1,1,0,4><<<gH2, 128, 0, stream>>>(HALF1, nullptr, u22w, u22b, HALF0, 64, HH, WH, HH, WH);
  bnrelu(HALF0, HALF0, u22g, u22bb, 64, HWH, 1, nullptr);           // e2 in HALF0
  conv_wmma_t<64,3,1,1,1,4><<<gF1, 128, 0, stream>>>(HALF0, nullptr, uupw, uupb, BEV0, 32, HF, WF, HF, WF);
  bnrelu(BEV0, BEV0, uupg, uupbb, 32, HWF, 1, nullptr);             // u in BEV0
  conv_wmma_t<64,3,1,1,2,4><<<gF1, 128, 0, stream>>>(BEV0, BEV2, ud1w, ud1b, BEV1, 32, HF, WF, HF, WF);
  bnrelu(BEV1, BEV1, ud1g, ud1bb, 32, HWF, 1, nullptr);             // d1 in BEV1
  conv_wmma_t<32,1,0,1,0,4><<<gF1, 128, 0, stream>>>(BEV1, nullptr, uow, uob, BEV2, 32, HF, WF, HF, WF);

  // 4) g2p + concat -> cmplt; sem MLP (WMMA for 64x64 layers)
  g2p_concat_kernel<<<blk(MPTS, T), T, 0, stream>>>(BEV2, FLAT, PW, CMPLT);
  gemm_wmma_t<64><<<dim3(MPTS / 64, 2), 128, 0, stream>>>(CMPLT, cw1, S1, 64, 1);
  gemm_wmma_t<64><<<dim3(MPTS / 64, 2), 128, 0, stream>>>(S1, cw2, S2, 64, 1);
  sem_head_kernel<<<blk(MPTS, T), T, 0, stream>>>(S2, cw3, SEM);

  // 5) selection: score + per-batch top-4096 (128 KB LDS bitonic sort)
  score_kernel<<<blk(MPTS, T), T, 0, stream>>>(SEM, SCORE);
  topk_kernel<<<dim3(BB), 1024, NPT * sizeof(unsigned long long), stream>>>(SCORE, SELROW);
  gather_sel_kernel<<<blk(BB * NSEL, T), T, 0, stream>>>(SELROW, pts, CMPLT, NEWFEAT, NFLAT);

  // 6) new_bev scatter-mean (64ch) + add out_bev into channels [0,32)
  hipMemsetAsync(BEV0, 0, (size_t)BB * 64 * HWF * sizeof(float), stream);
  hipMemsetAsync(CNT, 0, (size_t)BB * HWF * sizeof(float), stream);
  scatter_add_kernel<<<blk(BB * NSEL, T), T, 0, stream>>>(NEWFEAT, NFLAT, BEV0, CNT, BB * NSEL, 64);
  p2g_final_kernel<<<blk((long)BB * 64 * HWF, T), T, 0, stream>>>(BEV0, CNT, 64, BEV2, 32);

  // 7) basic block
  conv_wmma_t<64,3,1,1,0,4><<<gF2, 128, 0, stream>>>(BEV0, nullptr, bc1w, bc1b, BEV1, 64, HF, WF, HF, WF);
  bnrelu(BEV1, BEV1, bn1g, bn1b, 64, HWF, 1, nullptr);
  conv_wmma_t<64,3,1,1,0,4><<<gF2, 128, 0, stream>>>(BEV1, nullptr, bc2w, bc2b, BEV2, 64, HF, WF, HF, WF);
  bnrelu(BEV2, BEV2, bn2g, bn2b, 64, HWF, 1, BEV0);                 // bn + residual + relu
  conv_wmma_t<64,3,1,2,0,2><<<gS4, 64, 0, stream>>>(BEV2, nullptr, bc3w, bc3b, out, 128, HF, WF, HH, WH);
  bnrelu(out, out, bn3g, bn3b, 128, HWH, 1, nullptr);               // final [4,128,176,200]
}
